// HSTv4HyperLattice_26817775796988
// MI455X (gfx1250) — compile-verified, matrix operand order checked by reading, not run on hardware
//
#include <hip/hip_runtime.h>
#include <hip/hip_bf16.h>
#include <math.h>

// ---------------- model constants (must match reference) ----------------
#define Dm   512
#define Vm   32000
#define NHm  8
#define Lm   64
#define HORm 16
#define Bm   2
#define Sm   1024
#define HDm  64          // Dm / NHm
#define TOPKm 6          // int(Lm * 0.1)
#define EPSm 1e-5f

typedef __attribute__((ext_vector_type(16))) _Float16 v16h;
typedef __attribute__((ext_vector_type(8)))  float    v8f;

constexpr int ACT_NONE = 0, ACT_RELU = 1, ACT_GELU = 2, ACT_SIGMOID = 3;

// LDS pitches (in floats). 36/132 dwords: bank-conflict-avoiding and 16B-aligned.
#define PITCH_A  36      // A rows (64) and weight-B rows (128)
#define PITCH_P  132     // plain-B k-rows (32 x 128 floats)

// low 32 bits of a flat pointer into LDS == LDS byte offset (aperture truncation)
__device__ __forceinline__ unsigned lds_off(const void* p) {
  return (unsigned)(unsigned long long)p;
}

// async 16B copy global -> LDS (GLOBAL_LOAD_ASYNC_TO_LDS_B128, ASYNCcnt-tracked)
__device__ __forceinline__ void async_copy16(unsigned lds_byte, const float* gptr) {
  asm volatile("global_load_async_to_lds_b128 %0, %1, off"
               :: "v"(lds_byte), "v"(gptr)
               : "memory");
}

// ======================================================================
// Fragment assembly from LDS (CDNA5 ISA 7.12.2, 16-bit 16x32 operand, wave32)
// lanes 0-15 : row = lane,    e=0..7 -> K=e,     e=8..15 -> K=e+8
// lanes 16-31: row = lane-16, e=0..7 -> K=e+8,   e=8..15 -> K=e+16
// ======================================================================

// rows-along-K operand (A tile or (N,K) weight tile): elem (r,k) at base[r*pitch + k]
__device__ __forceinline__ v16h frag_lds_rows(const float* base, int pitch, int lane) {
  const int half = lane >> 4;
  const int r    = lane & 15;
  const float* q = base + r * pitch + 8 * half;
  v16h f;
#pragma unroll
  for (int e = 0; e < 8; ++e)  f[e] = (_Float16)q[e];
#pragma unroll
  for (int e = 8; e < 16; ++e) f[e] = (_Float16)q[e + 8];
  return f;
}

// plain (K,N) operand tile: elem (n,k) at base[k*pitch + n]
__device__ __forceinline__ v16h frag_lds_cols(const float* base, int pitch, int lane) {
  const int half = lane >> 4;
  const int n    = lane & 15;
  v16h f;
#pragma unroll
  for (int e = 0; e < 16; ++e) {
    const int k = e + 8 * half + ((e & 8) ? 8 : 0);
    f[e] = (_Float16)base[k * pitch + n];
  }
  return f;
}

// ======================================================================
// Batched WMMA GEMM:  C_z = act(A_z(M,K) @ B_z^T + bias) + residual
//   z = blockIdx.z; per-operand offset: (z/zdiv)*hi + (z%zdiv)*lo
//   B_IS_WEIGHT=true : B given as (N,K) row-major (PyTorch W, x @ W.T)
//   B_IS_WEIGHT=false: B given as (K,N) row-major (plain matmul)
// Block: 256 threads = 8 waves; block tile 64M x 128N x 32K, double-buffered
// async-to-LDS staging. Wave tile 16M x 64N: 4 B-fragments are materialized
// in distinct registers, then 4 WMMAs issue back-to-back (no WAR hazard NOPs).
// K must be a multiple of 32 (all uses: 64/128/512/1024/2048).
// ======================================================================
template <int ACT, bool B_IS_WEIGHT>
__global__ __launch_bounds__(256) void gemm_wmma(
    const float* __restrict__ A, long lda, long sAhi, long sAlo,
    const float* __restrict__ Bmat, long ldb, long sBhi, long sBlo,
    const float* __restrict__ bias,
    const float* __restrict__ resid, long ldr,
    float* __restrict__ C, long ldc, long sChi, long sClo,
    int zdiv, int M, int N, int K)
{
  __shared__ __align__(16) float sA[2][64  * PITCH_A];   //  9216 B each
  __shared__ __align__(16) float sB[2][128 * PITCH_A];   // 18432 B each (>= 32*PITCH_P)

  const int z  = blockIdx.z;
  const int zh = z / zdiv;
  const int zl = z % zdiv;
  A    += (long)zh * sAhi + (long)zl * sAlo;
  Bmat += (long)zh * sBhi + (long)zl * sBlo;
  C    += (long)zh * sChi + (long)zl * sClo;

  const int tid  = threadIdx.x;
  const int lane = tid & 31;
  const int w    = tid >> 5;           // 0..7
  const int wm   = w & 3;              // 4 m-tiles (16 rows each)
  const int wn   = w >> 2;             // 2 n-groups (64 cols each)
  const int bm   = blockIdx.y * 64;
  const int bn   = blockIdx.x * 128;
  const int mrem = M - bm;             // >= 1 by grid construction
  const int nrem = N - bn;             // >= 1

  v8f acc0, acc1, acc2, acc3;
#pragma unroll
  for (int i = 0; i < 8; ++i) { acc0[i] = 0.f; acc1[i] = 0.f; acc2[i] = 0.f; acc3[i] = 0.f; }

  // issue one k-step's staging: 6 async b128 copies per thread (2 A + 4 B)
  auto stage = [&](int buf, int k0) {
    // A tile: 64 rows x 32 f32; 512 transfers of 16B
#pragma unroll
    for (int j = 0; j < 2; ++j) {
      const int t    = tid + j * 256;        // 0..511
      const int row  = t >> 3;
      const int seg  = t & 7;
      const int rowc = row < mrem ? row : (mrem - 1);   // clamp: garbage masked at store
      const float* gp = A + (long)(bm + rowc) * lda + k0 + seg * 4;
      async_copy16(lds_off(&sA[buf][row * PITCH_A]) + seg * 16, gp);
    }
    if (B_IS_WEIGHT) {
      // B tile: 128 weight rows x 32 f32; 1024 transfers
#pragma unroll
      for (int j = 0; j < 4; ++j) {
        const int t    = tid + j * 256;      // 0..1023
        const int row  = t >> 3;
        const int seg  = t & 7;
        const int rowc = row < nrem ? row : (nrem - 1);
        const float* gp = Bmat + (long)(bn + rowc) * ldb + k0 + seg * 4;
        async_copy16(lds_off(&sB[buf][row * PITCH_A]) + seg * 16, gp);
      }
    } else {
      // B tile: 32 k-rows x 128 f32; 1024 transfers
#pragma unroll
      for (int j = 0; j < 4; ++j) {
        const int t   = tid + j * 256;       // 0..1023
        const int row = t >> 5;              // k row 0..31
        const int seg = t & 31;              // 16B segment within 512B row
        const float* gp = Bmat + (long)(k0 + row) * ldb + bn + seg * 4;
        async_copy16(lds_off(&sB[buf][row * PITCH_P]) + seg * 16, gp);
      }
    }
  };

  const int KT = K >> 5;
  stage(0, 0);
  for (int kt = 0; kt < KT; ++kt) {
    const int cur = kt & 1;
    if (kt + 1 < KT) {
      stage(cur ^ 1, (kt + 1) << 5);
      // 12 async ops outstanding; drain the 6 belonging to `cur` (in-order completion)
      asm volatile("s_wait_asynccnt 0x6" ::: "memory");
    } else {
      asm volatile("s_wait_asynccnt 0x0" ::: "memory");
    }
    __syncthreads();                    // cur buffer visible to all waves

    // materialize all fragments first (distinct registers), then 4 WMMAs
    const v16h af = frag_lds_rows(&sA[cur][(wm * 16) * PITCH_A], PITCH_A, lane);
    v16h bf0, bf1, bf2, bf3;
    if (B_IS_WEIGHT) {
      bf0 = frag_lds_rows(&sB[cur][(wn * 64 +  0) * PITCH_A], PITCH_A, lane);
      bf1 = frag_lds_rows(&sB[cur][(wn * 64 + 16) * PITCH_A], PITCH_A, lane);
      bf2 = frag_lds_rows(&sB[cur][(wn * 64 + 32) * PITCH_A], PITCH_A, lane);
      bf3 = frag_lds_rows(&sB[cur][(wn * 64 + 48) * PITCH_A], PITCH_A, lane);
    } else {
      bf0 = frag_lds_cols(&sB[cur][wn * 64 +  0], PITCH_P, lane);
      bf1 = frag_lds_cols(&sB[cur][wn * 64 + 16], PITCH_P, lane);
      bf2 = frag_lds_cols(&sB[cur][wn * 64 + 32], PITCH_P, lane);
      bf3 = frag_lds_cols(&sB[cur][wn * 64 + 48], PITCH_P, lane);
    }
    acc0 = __builtin_amdgcn_wmma_f32_16x16x32_f16(false, af, false, bf0, (short)0, acc0, false, false);
    acc1 = __builtin_amdgcn_wmma_f32_16x16x32_f16(false, af, false, bf1, (short)0, acc1, false, false);
    acc2 = __builtin_amdgcn_wmma_f32_16x16x32_f16(false, af, false, bf2, (short)0, acc2, false, false);
    acc3 = __builtin_amdgcn_wmma_f32_16x16x32_f16(false, af, false, bf3, (short)0, acc3, false, false);

    __syncthreads();                    // all waves done reading before restage
  }

  // ---- epilogue: VGPR v -> row wm*16 + v + 8*half, col tile + (lane&15) ----
  const int half = lane >> 4;
#pragma unroll
  for (int j = 0; j < 4; ++j) {
    const v8f& accj = (j == 0) ? acc0 : (j == 1) ? acc1 : (j == 2) ? acc2 : acc3;
    const int col = bn + wn * 64 + j * 16 + (lane & 15);
    if (col < N) {
#pragma unroll
      for (int v = 0; v < 8; ++v) {
        const int row = bm + wm * 16 + v + 8 * half;
        if (row < M) {
          float val = accj[v];
          if (bias) val += bias[col];
          if (ACT == ACT_RELU)         val = fmaxf(val, 0.f);
          else if (ACT == ACT_GELU)    val = 0.5f * val * (1.f + erff(val * 0.70710678118654752f));
          else if (ACT == ACT_SIGMOID) val = 1.f / (1.f + __expf(-val));
          if (resid) val += resid[(long)row * ldr + col];
          C[(long)row * ldc + col] = val;
        }
      }
    }
  }
}

// ======================================================================
// Elementwise / reduction kernels
// ======================================================================

__global__ __launch_bounds__(256) void embed_kernel(
    const int* __restrict__ ids, const float* __restrict__ tok,
    const float* __restrict__ pos, float* __restrict__ x)
{
  const long i = (long)blockIdx.x * 256 + threadIdx.x;   // over B*S*D
  const int  d = (int)(i % Dm);
  const long t = i / Dm;                                 // token index
  const int  s = (int)(t % Sm);
  x[i] = tok[(long)ids[t] * Dm + d] + pos[(long)s * Dm + d];
}

// out = LN(x [+ add]) * g + b     (one block per row, Dm = 512 = 2*256)
__global__ __launch_bounds__(256) void add_ln_kernel(
    const float* __restrict__ x, const float* __restrict__ add,
    const float* __restrict__ g, const float* __restrict__ bb,
    float* __restrict__ out)
{
  const int  row  = blockIdx.x;
  const long base = (long)row * Dm;
  const int  tid  = threadIdx.x;
  float e[2];
#pragma unroll
  for (int i = 0; i < 2; ++i) {
    const int d = tid + i * 256;
    float v = x[base + d];
    if (add) v += add[base + d];
    e[i] = v;
  }
  __shared__ float red[256];
  red[tid] = e[0] + e[1];
  __syncthreads();
  for (int o = 128; o > 0; o >>= 1) { if (tid < o) red[tid] += red[tid + o]; __syncthreads(); }
  const float mean = red[0] * (1.f / Dm);
  __syncthreads();
  red[tid] = (e[0] - mean) * (e[0] - mean) + (e[1] - mean) * (e[1] - mean);
  __syncthreads();
  for (int o = 128; o > 0; o >>= 1) { if (tid < o) red[tid] += red[tid + o]; __syncthreads(); }
  const float inv = rsqrtf(red[0] * (1.f / Dm) + EPSm);
#pragma unroll
  for (int i = 0; i < 2; ++i) {
    const int d = tid + i * 256;
    out[base + d] = (e[i] - mean) * inv * g[d] + bb[d];
  }
}

// batched row softmax with scale + optional causal mask; masked cols set to 0
// grid = (rows, nbatch); batch b's matrix starts at sc + b*bstride
__global__ __launch_bounds__(256) void softmax_kernel(
    float* __restrict__ sc, int ncols, long bstride, float scale, int causal)
{
  const int row   = blockIdx.x;
  float*    p     = sc + (long)blockIdx.y * bstride + (long)row * ncols;
  const int limit = causal ? (row + 1) : ncols;
  const int tid   = threadIdx.x;
  __shared__ float red[256];

  float mx = -3.4e38f;
  for (int c = tid; c < limit; c += 256) mx = fmaxf(mx, p[c] * scale);
  red[tid] = mx; __syncthreads();
  for (int o = 128; o > 0; o >>= 1) { if (tid < o) red[tid] = fmaxf(red[tid], red[tid + o]); __syncthreads(); }
  mx = red[0]; __syncthreads();

  float sum = 0.f;
  for (int c = tid; c < ncols; c += 256) {
    float e = 0.f;
    if (c < limit) { e = __expf(p[c] * scale - mx); sum += e; }
    p[c] = e;
  }
  red[tid] = sum; __syncthreads();
  for (int o = 128; o > 0; o >>= 1) { if (tid < o) red[tid] += red[tid + o]; __syncthreads(); }
  const float inv = 1.f / red[0];
  for (int c = tid; c < limit; c += 256) p[c] *= inv;
}

// top-6 of 64 router logits -> dense softmax gate row (one thread / token)
__global__ __launch_bounds__(256) void topk_gate_kernel(
    const float* __restrict__ rl, float* __restrict__ sd)
{
  const int t = blockIdx.x * 256 + threadIdx.x;
  if (t >= Bm * Sm) return;
  const float* r = rl + (long)t * Lm;
  int   ci[TOPKm];
  float cv[TOPKm];
#pragma unroll
  for (int j = 0; j < TOPKm; ++j) {
    float best = -3.4e38f; int bi = 0;
    for (int l = 0; l < Lm; ++l) {
      bool taken = false;
#pragma unroll
      for (int jj = 0; jj < TOPKm; ++jj)
        if (jj < j && ci[jj] == l) taken = true;
      const float v = r[l];
      if (!taken && v > best) { best = v; bi = l; }
    }
    ci[j] = bi; cv[j] = best;
  }
  const float mx = cv[0];
  float ex[TOPKm], sum = 0.f;
#pragma unroll
  for (int j = 0; j < TOPKm; ++j) { ex[j] = __expf(cv[j] - mx); sum += ex[j]; }
  const float inv = 1.f / sum;
  float* o = sd + (long)t * Lm;
  for (int l = 0; l < Lm; ++l) o[l] = 0.f;
#pragma unroll
  for (int j = 0; j < TOPKm; ++j) o[ci[j]] = ex[j] * inv;
}

// lattice_out[t,e] = sum_l sd[t,l] * sum_d x[t,d] * latW[l,d,e]
// one block per token; gate==0 skips (block-uniform branch)
__global__ __launch_bounds__(256) void lattice_kernel(
    const float* __restrict__ x, const float* __restrict__ lw,
    const float* __restrict__ sd, float* __restrict__ out)
{
  const int t   = blockIdx.x;
  const int tid = threadIdx.x;
  __shared__ float xs[Dm];
#pragma unroll
  for (int i = 0; i < 2; ++i) xs[tid + i * 256] = x[(long)t * Dm + tid + i * 256];
  __syncthreads();
  float acc0 = 0.f, acc1 = 0.f;
  const int e0 = tid, e1 = tid + 256;
  for (int l = 0; l < Lm; ++l) {
    const float wgt = sd[(long)t * Lm + l];
    if (wgt == 0.f) continue;
    const float* W = lw + (long)l * Dm * Dm;
    float a0 = 0.f, a1 = 0.f;
    for (int d = 0; d < Dm; ++d) {
      const float xv = xs[d];
      a0 += xv * W[(long)d * Dm + e0];
      a1 += xv * W[(long)d * Dm + e1];
    }
    acc0 += wgt * a0; acc1 += wgt * a1;
  }
  out[(long)t * Dm + e0] = acc0;
  out[(long)t * Dm + e1] = acc1;
}

// ======================================================================
// Host-side GEMM dispatch (non-batched convenience wrapper + batched)
// ======================================================================
static void launch_gemm_b(hipStream_t s,
                          const float* A, long lda, long sAhi, long sAlo,
                          const float* B, long ldb, long sBhi, long sBlo,
                          const float* bias, const float* resid, long ldr,
                          float* C, long ldc, long sChi, long sClo,
                          int zdiv, int nbatch, int M, int N, int K,
                          int act, bool b_is_weight)
{
  dim3 grid((unsigned)((N + 127) / 128), (unsigned)((M + 63) / 64), (unsigned)nbatch);
  dim3 blk(256);
  if (b_is_weight) {
    switch (act) {
      case ACT_RELU:
        gemm_wmma<ACT_RELU, true><<<grid, blk, 0, s>>>(A, lda, sAhi, sAlo, B, ldb, sBhi, sBlo,
            bias, resid, ldr, C, ldc, sChi, sClo, zdiv, M, N, K); break;
      case ACT_GELU:
        gemm_wmma<ACT_GELU, true><<<grid, blk, 0, s>>>(A, lda, sAhi, sAlo, B, ldb, sBhi, sBlo,
            bias, resid, ldr, C, ldc, sChi, sClo, zdiv, M, N, K); break;
      case ACT_SIGMOID:
        gemm_wmma<ACT_SIGMOID, true><<<grid, blk, 0, s>>>(A, lda, sAhi, sAlo, B, ldb, sBhi, sBlo,
            bias, resid, ldr, C, ldc, sChi, sClo, zdiv, M, N, K); break;
      default:
        gemm_wmma<ACT_NONE, true><<<grid, blk, 0, s>>>(A, lda, sAhi, sAlo, B, ldb, sBhi, sBlo,
            bias, resid, ldr, C, ldc, sChi, sClo, zdiv, M, N, K); break;
    }
  } else {
    gemm_wmma<ACT_NONE, false><<<grid, blk, 0, s>>>(A, lda, sAhi, sAlo, B, ldb, sBhi, sBlo,
        bias, resid, ldr, C, ldc, sChi, sClo, zdiv, M, N, K);
  }
}

static void launch_gemm(hipStream_t s,
                        const float* A, long lda, const float* B, long ldb,
                        const float* bias, const float* resid, long ldr,
                        float* C, long ldc, int M, int N, int K,
                        int act, bool b_is_weight)
{
  launch_gemm_b(s, A, lda, 0, 0, B, ldb, 0, 0, bias, resid, ldr,
                C, ldc, 0, 0, 1, 1, M, N, K, act, b_is_weight);
}

// ======================================================================
// kernel_launch
// ======================================================================
extern "C" void kernel_launch(void* const* d_in, const int* in_sizes, int n_in,
                              void* d_out, int out_size, void* d_ws, size_t ws_size,
                              hipStream_t stream)
{
  (void)in_sizes; (void)n_in; (void)out_size; (void)ws_size;

  // ---- inputs, flattened depth-first in setup_inputs() insertion order ----
  int i = 0;
  const int*   input_ids = (const int*)d_in[i++];
  const float* tok_emb   = (const float*)d_in[i++];
  const float* pos_emb   = (const float*)d_in[i++];

  struct BotL { const float *Wq,*bq,*Wk,*bk,*Wv,*bv,*Wo,*bo,*W1,*b1,*W2,*b2,*l1g,*l1b,*l2g,*l2b; } bot[2];
  for (int l = 0; l < 2; ++l) {
    bot[l].Wq  = (const float*)d_in[i++]; bot[l].bq  = (const float*)d_in[i++];
    bot[l].Wk  = (const float*)d_in[i++]; bot[l].bk  = (const float*)d_in[i++];
    bot[l].Wv  = (const float*)d_in[i++]; bot[l].bv  = (const float*)d_in[i++];
    bot[l].Wo  = (const float*)d_in[i++]; bot[l].bo  = (const float*)d_in[i++];
    bot[l].W1  = (const float*)d_in[i++]; bot[l].b1  = (const float*)d_in[i++];
    bot[l].W2  = (const float*)d_in[i++]; bot[l].b2  = (const float*)d_in[i++];
    bot[l].l1g = (const float*)d_in[i++]; bot[l].l1b = (const float*)d_in[i++];
    bot[l].l2g = (const float*)d_in[i++]; bot[l].l2b = (const float*)d_in[i++];
  }
  const float* gate_W    = (const float*)d_in[i++];
  const float* lattice_W = (const float*)d_in[i++];
  const float* lat_out_W = (const float*)d_in[i++];
  const float* lat_out_b = (const float*)d_in[i++];
  const float* lat_ln_g  = (const float*)d_in[i++];
  const float* lat_ln_b  = (const float*)d_in[i++];

  struct TopL { const float *Wq,*Wk,*Wv,*Wo,*W1,*b1,*W2,*b2,*l1g,*l1b,*l2g,*l2b; } topl[2];
  for (int l = 0; l < 2; ++l) {
    topl[l].Wq  = (const float*)d_in[i++]; topl[l].Wk  = (const float*)d_in[i++];
    topl[l].Wv  = (const float*)d_in[i++]; topl[l].Wo  = (const float*)d_in[i++];
    topl[l].W1  = (const float*)d_in[i++]; topl[l].b1  = (const float*)d_in[i++];
    topl[l].W2  = (const float*)d_in[i++]; topl[l].b2  = (const float*)d_in[i++];
    topl[l].l1g = (const float*)d_in[i++]; topl[l].l1b = (const float*)d_in[i++];
    topl[l].l2g = (const float*)d_in[i++]; topl[l].l2b = (const float*)d_in[i++];
  }
  const float* lnf_g      = (const float*)d_in[i++];
  const float* lnf_b      = (const float*)d_in[i++];
  const float* lm_W       = (const float*)d_in[i++];
  const float* hor_W      = (const float*)d_in[i++];
  const float* hor_b      = (const float*)d_in[i++];
  const float* hor_pred_W = (const float*)d_in[i++];
  const float* c1_W       = (const float*)d_in[i++];
  const float* c1_b       = (const float*)d_in[i++];
  const float* c2_W       = (const float*)d_in[i++];
  const float* c2_b       = (const float*)d_in[i++];

  // ---- workspace carve (~45 MB floats) ----
  const long BSD = (long)Bm * Sm * Dm;
  float* ws  = (float*)d_ws;
  float* X   = ws;  ws += BSD;                 // activations
  float* T1  = ws;  ws += BSD;                 // scratch (attn out / ffn out)
  float* T2  = ws;  ws += BSD;                 // scratch (nx / pre-LN)
  float* Qb  = ws;  ws += BSD;
  float* Kb  = ws;  ws += BSD;
  float* Vb  = ws;  ws += BSD;
  float* HID = ws;  ws += (long)Bm * Sm * 4 * Dm;
  float* RL  = ws;  ws += (long)Bm * Sm * Lm;
  float* SD  = ws;  ws += (long)Bm * Sm * Lm;
  float* LAT = ws;  ws += BSD;
  float* PRJ = ws;  ws += (long)Bm * HORm * Dm;
  float* CH  = ws;  ws += (long)Bm * (Dm / 4);

  float* out     = (float*)d_out;
  float* logits  = out;                              // (B,S,V)
  float* horiz   = logits + (long)Bm * Sm * Vm;      // (B,HOR,V)
  float* conf    = horiz  + (long)Bm * HORm * Vm;    // (B,HOR)

  // Attention score scratch (B*NH*S*S = 16.8M floats) lives in the logits
  // region of d_out (65.5M floats) -- it is dead until the final LM-head GEMM,
  // which fully overwrites it.  Deterministic and graph-capture safe.
  float* SCb = logits;

  const int  M_  = Bm * Sm;
  const int  BH  = Bm * NHm;
  const long SS  = (long)Sm * Sm;
  const long SDl = (long)Sm * Dm;

  // ---- multi-head attention, all B*NH head-slices batched via blockIdx.z ----
  auto attn = [&](const float* xin,
                  const float* Wq, const float* bq,
                  const float* Wk, const float* bk,
                  const float* Wv, const float* bv,
                  int causal, float* oout) {
    launch_gemm(stream, xin, Dm, Wq, Dm, bq, nullptr, 0, Qb, Dm, M_, Dm, Dm, ACT_NONE, true);
    launch_gemm(stream, xin, Dm, Wk, Dm, bk, nullptr, 0, Kb, Dm, M_, Dm, Dm, ACT_NONE, true);
    launch_gemm(stream, xin, Dm, Wv, Dm, bv, nullptr, 0, Vb, Dm, M_, Dm, Dm, ACT_NONE, true);
    // scores_z = Q_z @ K_z^T ; z=(b,h), Q/K offset = b*S*D + h*64, SC offset = z*S*S
    launch_gemm_b(stream,
                  Qb, Dm, SDl, HDm,
                  Kb, Dm, SDl, HDm,
                  nullptr, nullptr, 0,
                  SCb, Sm, (long)NHm * SS, SS,
                  NHm, BH, Sm, Sm, HDm, ACT_NONE, true);
    softmax_kernel<<<dim3(Sm, BH), 256, 0, stream>>>(SCb, Sm, SS, 0.125f /*1/sqrt(64)*/, causal);
    // o_z = P_z @ V_z   (V slice is plain (K=S, N=hd))
    launch_gemm_b(stream,
                  SCb, Sm, (long)NHm * SS, SS,
                  Vb, Dm, SDl, HDm,
                  nullptr, nullptr, 0,
                  oout, Dm, SDl, HDm,
                  NHm, BH, Sm, HDm, Sm, ACT_NONE, false);
  };

  // ---- embedding ----
  embed_kernel<<<(unsigned)(BSD / 256), 256, 0, stream>>>(input_ids, tok_emb, pos_emb, X);

  // ---- bottom layers (post-LN, ReLU FFN, biased) ----
  for (int l = 0; l < 2; ++l) {
    const BotL& p = bot[l];
    attn(X, p.Wq, p.bq, p.Wk, p.bk, p.Wv, p.bv, /*causal=*/0, T1);
    launch_gemm(stream, T1, Dm, p.Wo, Dm, p.bo, nullptr, 0, T2, Dm, M_, Dm, Dm, ACT_NONE, true);
    add_ln_kernel<<<M_, 256, 0, stream>>>(X, T2, p.l1g, p.l1b, X);
    launch_gemm(stream, X, Dm, p.W1, Dm, p.b1, nullptr, 0, HID, 4 * Dm, M_, 4 * Dm, Dm, ACT_RELU, true);
    launch_gemm(stream, HID, 4 * Dm, p.W2, 4 * Dm, p.b2, nullptr, 0, T1, Dm, M_, Dm, 4 * Dm, ACT_NONE, true);
    add_ln_kernel<<<M_, 256, 0, stream>>>(X, T1, p.l2g, p.l2b, X);
  }

  // ---- router + sparse lattice mixture ----
  launch_gemm(stream, X, Dm, gate_W, Dm, nullptr, nullptr, 0, RL, Lm, M_, Lm, Dm, ACT_NONE, true);
  topk_gate_kernel<<<(M_ + 255) / 256, 256, 0, stream>>>(RL, SD);
  lattice_kernel<<<M_, 256, 0, stream>>>(X, lattice_W, SD, LAT);
  launch_gemm(stream, LAT, Dm, lat_out_W, Dm, lat_out_b, nullptr, 0, T1, Dm, M_, Dm, Dm, ACT_NONE, true);
  add_ln_kernel<<<M_, 256, 0, stream>>>(X, T1, lat_ln_g, lat_ln_b, X);

  // ---- top layers (pre-LN, causal, exact GELU FFN, unbiased QKV) ----
  for (int l = 0; l < 2; ++l) {
    const TopL& p = topl[l];
    add_ln_kernel<<<M_, 256, 0, stream>>>(X, nullptr, p.l1g, p.l1b, T2);       // nx
    attn(T2, p.Wq, nullptr, p.Wk, nullptr, p.Wv, nullptr, /*causal=*/1, T1);
    launch_gemm(stream, T1, Dm, p.Wo, Dm, nullptr, X, Dm, X, Dm, M_, Dm, Dm, ACT_NONE, true);  // x += o@Wo^T
    add_ln_kernel<<<M_, 256, 0, stream>>>(X, nullptr, p.l2g, p.l2b, T2);       // n2
    launch_gemm(stream, T2, Dm, p.W1, Dm, p.b1, nullptr, 0, HID, 4 * Dm, M_, 4 * Dm, Dm, ACT_GELU, true);
    launch_gemm(stream, HID, 4 * Dm, p.W2, 4 * Dm, p.b2, X, Dm, X, Dm, M_, Dm, 4 * Dm, ACT_NONE, true);
  }

  // ---- LM head (overwrites the score scratch region of d_out) ----
  add_ln_kernel<<<M_, 256, 0, stream>>>(X, nullptr, lnf_g, lnf_b, T2);
  launch_gemm(stream, T2, Dm, lm_W, Dm, nullptr, nullptr, 0, logits, Vm, M_, Vm, Dm, ACT_NONE, true);

  // ---- horizon + confidence heads on x_last (strided rows, lda = S*D) ----
  const float* xlast = X + (long)(Sm - 1) * Dm;
  launch_gemm(stream, xlast, SDl, hor_W, Dm, hor_b, nullptr, 0,
              PRJ, HORm * Dm, Bm, HORm * Dm, Dm, ACT_NONE, true);
  launch_gemm(stream, PRJ, Dm, hor_pred_W, Dm, nullptr, nullptr, 0,
              horiz, Vm, Bm * HORm, Vm, Dm, ACT_NONE, true);
  launch_gemm(stream, xlast, SDl, c1_W, Dm, c1_b, nullptr, 0,
              CH, Dm / 4, Bm, Dm / 4, Dm, ACT_RELU, true);
  launch_gemm(stream, CH, Dm / 4, c2_W, Dm / 4, c2_b, nullptr, 0,
              conf, HORm, Bm, HORm, Dm / 4, ACT_SIGMOID, true);
}
